// PAttention_82798379532414
// MI455X (gfx1250) — matrix operand order repaired and checked by reading, not compiled, not run on hardware
//
#include <hip/hip_runtime.h>
#include <hip/hip_bf16.h>

// ---- CDNA5 (gfx1250) wave32 WMMA types ----
typedef __attribute__((ext_vector_type(16))) __bf16 v16bf;
typedef __attribute__((ext_vector_type(8)))  __bf16 v8bf;
typedef __attribute__((ext_vector_type(8)))  float  v8f;
typedef __attribute__((ext_vector_type(16))) unsigned short v16us;
typedef __attribute__((ext_vector_type(4)))  float  f32x4;
typedef __attribute__((ext_vector_type(4)))  unsigned int u32x4;
typedef __attribute__((ext_vector_type(8)))  int i32x8;
typedef __attribute__((ext_vector_type(4)))  int i32x4;

static constexpr int L  = 4096;   // query length
static constexpr int D  = 2048;   // head dim
static constexpr int NT = 4096;   // n_tokens (key/value bank)
static constexpr int W  = 2304;   // padded band width (>= 2063 needed; 16 waves * 9 tiles * 16)
static constexpr float SCALE = 0.02209708691207961f; // 1/sqrt(2048)

// f32 -> bf16, round-to-nearest-even (pure VALU)
__device__ __forceinline__ unsigned short f2bf(float f) {
  unsigned u = __builtin_bit_cast(unsigned, f);
  u += 0x7FFFu + ((u >> 16) & 1u);
  return (unsigned short)(u >> 16);
}

__device__ __forceinline__ v16bf pack16(f32x4 a, f32x4 b, f32x4 c, f32x4 d) {
  v16us u;
#pragma unroll
  for (int i = 0; i < 4; ++i) {
    u[i]      = f2bf(a[i]);
    u[4 + i]  = f2bf(b[i]);
    u[8 + i]  = f2bf(c[i]);
    u[12 + i] = f2bf(d[i]);
  }
  return __builtin_bit_cast(v16bf, u);
}

__device__ __forceinline__ v16bf concat8(v8bf lo, v8bf hi) {
  return __builtin_shufflevector(lo, hi, 0,1,2,3,4,5,6,7,8,9,10,11,12,13,14,15);
}

// ---- Tensor Data Mover: DMA a 2D row-tile of bf16 into LDS (padded) ----
// D# per CDNA5 ISA. data_size=2B; LDS padding: 4 DWORDs
// (16B) inserted every 64 DWORDs (256B) so staged rows rotate across banks
// while 16B lane chunks stay 16B-aligned. Consumers must apply
//   padded(o) = o + ((o >> 8) << 4)
__device__ __forceinline__ void tdm_load_tile_to_lds(const void* gsrc,
                                                     unsigned lds_byte_off,
                                                     unsigned row_stride_elems,
                                                     unsigned tile_cols,
                                                     unsigned tile_rows) {
  unsigned long long ga = (unsigned long long)(uintptr_t)gsrc;
  u32x4 g0;
  g0[0] = 1u;                                                    // count=1, user D#
  g0[1] = lds_byte_off;                                          // lds_addr
  g0[2] = (unsigned)ga;                                          // global_addr[31:0]
  g0[3] = ((unsigned)(ga >> 32) & 0x01FFFFFFu) | (2u << 30);     // addr[56:32] | type=2
  i32x8 g1;
  g1[0] = (1 << 16) | (1 << 20) | (5 << 22) | (3 << 25);         // data_size=2B, pad_en, interval=64DW, amount=4DW
  g1[1] = (int)((row_stride_elems & 0xFFFFu) << 16);             // tensor_dim0[15:0]
  g1[2] = (int)(((row_stride_elems >> 16) & 0xFFFFu) |
                ((tile_rows & 0xFFFFu) << 16));                  // dim0[31:16] | tensor_dim1[15:0]
  g1[3] = (int)((tile_cols & 0xFFFFu) << 16);                    // tensor_dim1[31:16]=0 | tile_dim0
  g1[4] = (int)(tile_rows & 0xFFFFu);                            // tile_dim1 | tile_dim2=0
  g1[5] = (int)row_stride_elems;                                 // tensor_dim0_stride[31:0]
  g1[6] = 0;
  g1[7] = 0;
  i32x4 z4 = (i32x4)0;
#if defined(__clang_major__) && __clang_major__ >= 23
  i32x8 z8 = (i32x8)0;
  __builtin_amdgcn_tensor_load_to_lds(g0, g1, z4, z4, z8, 0);
#else
  __builtin_amdgcn_tensor_load_to_lds(g0, g1, z4, z4, 0);
#endif
}

// ---------------- prep: f32 -> bf16 elementwise (Pk -> Kb) ----------------
__global__ void pattn_cvt_bf16(const float* __restrict__ src,
                               unsigned short* __restrict__ dst, int n4) {
  int i = blockIdx.x * blockDim.x + threadIdx.x;
  if (i >= n4) return;
  f32x4 v = reinterpret_cast<const f32x4*>(src)[i];
  unsigned long long p = (unsigned long long)f2bf(v[0])
                       | ((unsigned long long)f2bf(v[1]) << 16)
                       | ((unsigned long long)f2bf(v[2]) << 32)
                       | ((unsigned long long)f2bf(v[3]) << 48);
  reinterpret_cast<unsigned long long*>(dst)[i] = p;
}

// ---------------- prep: Pv (NT x D) f32 -> VbT (D x NT) bf16 ----------------
__global__ void pattn_transpose_bf16(const float* __restrict__ pv,
                                     unsigned short* __restrict__ vt) {
  __shared__ float tile[32][33];
  const int c0 = blockIdx.x * 32;  // d index
  const int t0 = blockIdx.y * 32;  // token index
  const int tx = threadIdx.x, ty = threadIdx.y; // 32 x 8
#pragma unroll
  for (int r = ty; r < 32; r += 8)
    tile[r][tx] = pv[(size_t)(t0 + r) * D + c0 + tx];
  __syncthreads();
#pragma unroll
  for (int r = ty; r < 32; r += 8)
    vt[(size_t)(c0 + r) * NT + t0 + tx] = f2bf(tile[tx][r]);
}

// ---------------- Q projection: Qb = bf16(X @ Wq^T) ----------------
// grid (L/16, D/128), 256 threads (8 waves), wave w -> 16x16 tile at col0+16w
__global__ void __launch_bounds__(256, 1)
pattn_qproj(const float* __restrict__ x, const float* __restrict__ wq,
            unsigned short* __restrict__ qb) {
  const int wave = threadIdx.x >> 5;
  const int lane = threadIdx.x & 31;
  const int n = lane & 15;
  const int h = lane >> 4;
  const int row0 = blockIdx.x * 16;
  const int col0 = blockIdx.y * 128 + wave * 16;

  const float* ar = x  + (size_t)(row0 + n) * D;   // A: lane row m = n
  const float* br = wq + (size_t)(col0 + n) * D;   // B: lane col n -> Wq row (contiguous over k)
  v8f acc = (v8f)0.0f;
  for (int k = 0; k < D; k += 32) {
    const float* ap = ar + k + 16 * h;
    f32x4 a0 = *reinterpret_cast<const f32x4*>(ap);
    f32x4 a1 = *reinterpret_cast<const f32x4*>(ap + 4);
    f32x4 a2 = *reinterpret_cast<const f32x4*>(ap + 8);
    f32x4 a3 = *reinterpret_cast<const f32x4*>(ap + 12);
    v16bf A = pack16(a0, a1, a2, a3);
    const float* bp = br + k + 16 * h;
    f32x4 b0 = *reinterpret_cast<const f32x4*>(bp);
    f32x4 b1 = *reinterpret_cast<const f32x4*>(bp + 4);
    f32x4 b2 = *reinterpret_cast<const f32x4*>(bp + 8);
    f32x4 b3 = *reinterpret_cast<const f32x4*>(bp + 12);
    v16bf B = pack16(b0, b1, b2, b3);
    acc = __builtin_amdgcn_wmma_f32_16x16x32_bf16(false, A, false, B,
                                                  (short)0, acc, false, false);
  }
#pragma unroll
  for (int g = 0; g < 8; ++g)
    qb[(size_t)(row0 + 8 * h + g) * D + col0 + n] = f2bf(acc[g]);
}

// ---------------- banded scores + exact f32 softmax -> Pb (bf16) ----------------
// grid (L/16), 512 threads (16 waves); wave w owns 9 column tiles of the W band.
// The shared 16-row Q tile is staged into LDS by the TDM in two segments.
__global__ void __launch_bounds__(512, 1)
pattn_scores_softmax(const unsigned short* __restrict__ qb,
                     const unsigned short* __restrict__ kb,
                     unsigned short* __restrict__ pb,
                     const int* __restrict__ winp) {
  constexpr int NTILE = 9;
  constexpr int SEG   = 1024;                 // k elems per staged segment
  constexpr int LROW  = 2176;                 // padded LDS row stride (2048 + 8*16)
  __shared__ struct {
    __align__(16) char qtile[16 * LROW];
    float red[16][16];
    float stat[16];
  } sm;

  const int wave = threadIdx.x >> 5;
  const int lane = threadIdx.x & 31;
  const int n = lane & 15;
  const int h = lane >> 4;
  const int l0 = blockIdx.x * 16;
  int bbase = l0 - 1024;
  bbase = bbase < 0 ? 0 : bbase;
  bbase = bbase > (NT - W) ? (NT - W) : bbase;
  const int wcol = wave * (NTILE * 16);
  const int win = *winp;

  v8f acc[NTILE];
#pragma unroll
  for (int ct = 0; ct < NTILE; ++ct) acc[ct] = (v8f)0.0f;

  const char* arow = sm.qtile + n * LROW;

  for (int seg = 0; seg < 2; ++seg) {
    __syncthreads();                          // previous segment fully consumed
    if (wave == 0) {
      tdm_load_tile_to_lds(qb + (size_t)l0 * D + seg * SEG,
                           (unsigned)(uintptr_t)sm.qtile, D, SEG, 16);
      __builtin_amdgcn_s_wait_tensorcnt(0);
    }
    __syncthreads();                          // tile visible to all waves
    for (int kk = 0; kk < SEG; kk += 32) {
      int o0 = (kk + 16 * h) * 2;
      v8bf alo = *reinterpret_cast<const v8bf*>(arow + o0 + ((o0 >> 8) << 4));
      int o1 = o0 + 16;
      v8bf ahi = *reinterpret_cast<const v8bf*>(arow + o1 + ((o1 >> 8) << 4));
      v16bf A = concat8(alo, ahi);
      const unsigned short* brow =
          kb + (size_t)(bbase + wcol + n) * D + seg * SEG + kk + 16 * h;
      __builtin_prefetch(brow + 32, 0, 1);
#pragma unroll
      for (int ct = 0; ct < NTILE; ++ct) {
        v16bf B = *reinterpret_cast<const v16bf*>(brow + (size_t)ct * 16 * D);
        acc[ct] = __builtin_amdgcn_wmma_f32_16x16x32_bf16(false, A, false, B,
                                                          (short)0, acc[ct], false, false);
      }
    }
  }

  // scale + sliding-window mask (in-register, full band is resident)
#pragma unroll
  for (int ct = 0; ct < NTILE; ++ct) {
    const int t = bbase + wcol + ct * 16 + n;
#pragma unroll
    for (int g = 0; g < 8; ++g) {
      const int l = l0 + 8 * h + g;
      int dlt = l - t; dlt = dlt < 0 ? -dlt : dlt;
      float s = acc[ct][g] * SCALE;
      acc[ct][g] = (dlt < win) ? s : -3.0e38f;
    }
  }

  // ---- row max ----
  float pm[8];
#pragma unroll
  for (int g = 0; g < 8; ++g) {
    float m = -3.0e38f;
#pragma unroll
    for (int ct = 0; ct < NTILE; ++ct) m = fmaxf(m, acc[ct][g]);
#pragma unroll
    for (int s = 1; s < 16; s <<= 1) m = fmaxf(m, __shfl_xor(m, s, 16));
    pm[g] = m;
  }
  if (n == 0) {
#pragma unroll
    for (int g = 0; g < 8; ++g) sm.red[wave][8 * h + g] = pm[g];
  }
  __syncthreads();
  if (threadIdx.x < 16) {
    float m = sm.red[0][threadIdx.x];
#pragma unroll
    for (int w2 = 1; w2 < 16; ++w2) m = fmaxf(m, sm.red[w2][threadIdx.x]);
    sm.stat[threadIdx.x] = m;
  }
  __syncthreads();
  float rmax[8];
#pragma unroll
  for (int g = 0; g < 8; ++g) rmax[g] = sm.stat[8 * h + g];
  __syncthreads();

  // ---- exp + row sum ----
  float ps[8];
#pragma unroll
  for (int g = 0; g < 8; ++g) {
    float s = 0.0f;
#pragma unroll
    for (int ct = 0; ct < NTILE; ++ct) {
      float e = __expf(acc[ct][g] - rmax[g]);
      acc[ct][g] = e;
      s += e;
    }
#pragma unroll
    for (int st = 1; st < 16; st <<= 1) s += __shfl_xor(s, st, 16);
    ps[g] = s;
  }
  if (n == 0) {
#pragma unroll
    for (int g = 0; g < 8; ++g) sm.red[wave][8 * h + g] = ps[g];
  }
  __syncthreads();
  if (threadIdx.x < 16) {
    float s = sm.red[0][threadIdx.x];
#pragma unroll
    for (int w2 = 1; w2 < 16; ++w2) s += sm.red[w2][threadIdx.x];
    sm.stat[threadIdx.x] = s;
  }
  __syncthreads();
  float rinv[8];
#pragma unroll
  for (int g = 0; g < 8; ++g) rinv[g] = 1.0f / sm.stat[8 * h + g];

  // ---- store probabilities (masked entries are exactly 0 -> PV needs no mask) ----
#pragma unroll
  for (int ct = 0; ct < NTILE; ++ct) {
#pragma unroll
    for (int g = 0; g < 8; ++g)
      pb[(size_t)(l0 + 8 * h + g) * W + wcol + ct * 16 + n] =
          f2bf(acc[ct][g] * rinv[g]);
  }
}

// ---------------- y = Pb @ Pv over the band (via VbT) ----------------
// grid (L/16, D/128), 256 threads; wave w -> 16x16 output tile.
// Shared 16-row Pb tile staged via TDM (row stride 2448B -> bank-conflict-free).
__global__ void __launch_bounds__(256, 1)
pattn_pv(const unsigned short* __restrict__ pb,
         const unsigned short* __restrict__ vt,
         float* __restrict__ y) {
  constexpr int SEGV  = 1152;                 // band elems per staged segment
  constexpr int LROWV = 2448;                 // padded LDS row stride (2304 + 9*16)
  __shared__ struct {
    __align__(16) char ptile[16 * LROWV];
  } sm;

  const int wave = threadIdx.x >> 5;
  const int lane = threadIdx.x & 31;
  const int n = lane & 15;
  const int h = lane >> 4;
  const int l0 = blockIdx.x * 16;
  const int col0 = blockIdx.y * 128 + wave * 16;
  int bbase = l0 - 1024;
  bbase = bbase < 0 ? 0 : bbase;
  bbase = bbase > (NT - W) ? (NT - W) : bbase;

  const char* arow = sm.ptile + n * LROWV;
  const unsigned short* bp = vt + (size_t)(col0 + n) * NT + bbase + 16 * h;
  v8f acc = (v8f)0.0f;

  for (int seg = 0; seg < 2; ++seg) {
    __syncthreads();
    if (wave == 0) {
      tdm_load_tile_to_lds(pb + (size_t)l0 * W + seg * SEGV,
                           (unsigned)(uintptr_t)sm.ptile, W, SEGV, 16);
      __builtin_amdgcn_s_wait_tensorcnt(0);
    }
    __syncthreads();
    for (int kk = 0; kk < SEGV; kk += 32) {
      int o0 = (kk + 16 * h) * 2;
      v8bf alo = *reinterpret_cast<const v8bf*>(arow + o0 + ((o0 >> 8) << 4));
      int o1 = o0 + 16;
      v8bf ahi = *reinterpret_cast<const v8bf*>(arow + o1 + ((o1 >> 8) << 4));
      v16bf A = concat8(alo, ahi);
      v16bf B = *reinterpret_cast<const v16bf*>(bp + seg * SEGV + kk);
      __builtin_prefetch(bp + seg * SEGV + kk + 64, 0, 1);
      acc = __builtin_amdgcn_wmma_f32_16x16x32_bf16(false, A, false, B,
                                                    (short)0, acc, false, false);
    }
  }
#pragma unroll
  for (int g = 0; g < 8; ++g)
    y[(size_t)(l0 + 8 * h + g) * D + col0 + n] = acc[g];
}

extern "C" void kernel_launch(void* const* d_in, const int* in_sizes, int n_in,
                              void* d_out, int out_size, void* d_ws, size_t ws_size,
                              hipStream_t stream) {
  (void)in_sizes; (void)n_in; (void)out_size; (void)ws_size;
  const float* x   = (const float*)d_in[0];   // (L, D)
  const float* wq  = (const float*)d_in[1];   // (D, D)
  const float* pk  = (const float*)d_in[2];   // (1, NT, D)
  const float* pv  = (const float*)d_in[3];   // (1, NT, D)
  const int*   win = (const int*)d_in[4];     // scalar
  float* out = (float*)d_out;                 // (1, L, D) f32

  unsigned short* qb  = (unsigned short*)d_ws;         // L*D   bf16
  unsigned short* kb  = qb + (size_t)L * D;            // NT*D  bf16
  unsigned short* vtp = kb + (size_t)NT * D;           // D*NT  bf16 (Pv transposed)
  unsigned short* pbp = vtp + (size_t)D * NT;          // L*W   bf16 (banded probs)

  // prep: Pk -> bf16
  {
    int n4 = NT * D / 4;
    pattn_cvt_bf16<<<(n4 + 255) / 256, 256, 0, stream>>>(pk, kb, n4);
  }
  // prep: Pv -> bf16 transposed
  {
    dim3 tb(32, 8), tg(D / 32, NT / 32);
    pattn_transpose_bf16<<<tg, tb, 0, stream>>>(pv, vtp);
  }
  // Q projection
  pattn_qproj<<<dim3(L / 16, D / 128), 256, 0, stream>>>(x, wq, qb);
  // banded scores + softmax (TDM-staged Q tile)
  pattn_scores_softmax<<<L / 16, 512, 0, stream>>>(qb, kb, pbp, win);
  // PV product (TDM-staged Pb tile)
  pattn_pv<<<dim3(L / 16, D / 128), 256, 0, stream>>>(pbp, vtp, out);
}
